// BiDirSAGEClassifier_9552007266357
// MI455X (gfx1250) — matrix operand order repaired and checked by reading, not compiled, not run on hardware
//
#include <hip/hip_runtime.h>
#include <hip/hip_bf16.h>

#define N_NODES 262144
#define N_EDGES 524288
#define FT 300
#define HD 128
#define NLAYERS 3
#define NUMF 16
#define BN_EPS 1e-5f

typedef __attribute__((ext_vector_type(16))) _Float16 v16h;
typedef __attribute__((ext_vector_type(8)))  _Float16 f16x8;
typedef __attribute__((ext_vector_type(8)))  float    v8f;

// ---------------------------------------------------------------------------
// Weight prep: f32 [K x 128] row-major -> f16 in B-fragment-linear layout:
//   out[((kt*8 + c)*32 + lane)*16 + j] = W[kt*32 + (lane>>4)*8 + j + (j>=8)*8][c*16 + (lane&15)]
// so the GEMM loads each 32x16 B fragment as two coalesced global_load_b128.
// Zero-padded when k >= K. Total size = ceil(K/32)*4096 halves.
// ---------------------------------------------------------------------------
__global__ __launch_bounds__(256) void prep_weight_kernel(
    const float* __restrict__ W, int K, _Float16* __restrict__ out, int total)
{
    const int t = blockIdx.x * 256 + threadIdx.x;
    if (t >= total) return;
    const int j    = t & 15;
    const int lane = (t >> 4) & 31;
    const int c    = (t >> 9) & 7;
    const int kt   = t >> 12;
    const int k = kt * 32 + (lane >> 4) * 8 + j + ((j >= 8) ? 8 : 0);
    const int n = c * 16 + (lane & 15);
    out[t] = (k < K) ? (_Float16)W[k * 128 + n] : (_Float16)0.0f;
}

__device__ __forceinline__ void load_bfrag(v16h& b, const _Float16* bp)
{
    *(f16x8*)&b         = *(const f16x8*)bp;
    *(((f16x8*)&b) + 1) = *(const f16x8*)(bp + 8);
}

// ---------------------------------------------------------------------------
// Fused GEMM: out[N x 128] = epilogue( A0 @ W0 + (DUAL ? A1 @ W1 : 0) )
//   A*: f32 row-major [N x K*]; A0 gathered through g0 if GATHER.
//   Wp*: f16 weights pre-swizzled by prep_weight_kernel (L2-resident).
//   MODE 0: acc+bias | 1: relu(acc+bias) | 2: relu(relu(acc+bias)+addb)
// 256 threads = 8 waves; block tile 128x128; wave tile 16x128 (8 accumulators).
// No LDS, no barriers: A fragments are built straight from global f32 with
// four aligned float4 loads per lane (lanes m and m+16 jointly consume one
// contiguous 128B line per row per k-step), converted to f16 in-register;
// B fragments are two global_load_b128 from the pre-swizzled f16 weights,
// double-buffered so fragment c+1 streams in while WMMA c executes.
// Tail k-tiles (K=300, K=16) are branch-free via clamped loads + select.
// ---------------------------------------------------------------------------
template <int MODE, bool DUAL, bool GATHER>
__global__ __launch_bounds__(256) void gemm128_wmma(
    const float* __restrict__ A0, const int* __restrict__ g0,
    const _Float16* __restrict__ Wp0, int K0,
    const float* __restrict__ A1, const _Float16* __restrict__ Wp1, int K1,
    const float* __restrict__ bias, const float* __restrict__ addb,
    float* __restrict__ out)
{
    const int tid  = threadIdx.x;
    const int lane = tid & 31;
    const int wave = tid >> 5;
    const int m    = lane & 15;
    const int hi   = lane >> 4;
    const long rowBase = (long)blockIdx.x * 128;
    const long myRow   = rowBase + wave * 16 + m;   // A row owned by this lane

    v8f acc[8] = {};

    const int nsrc = DUAL ? 2 : 1;
    for (int s = 0; s < nsrc; ++s) {
        const float*    A  = (DUAL && s) ? A1 : A0;
        const _Float16* Wp = (DUAL && s) ? Wp1 : Wp0;
        const int       K  = (DUAL && s) ? K1 : K0;
        const int ktiles = (K + 31) >> 5;
        const int kfull  = K >> 5;

        const long row = (GATHER && s == 0) ? (long)g0[myRow] : myRow;
        const float* arow = A + row * K;  // 16B-aligned: K in {16,128,300}

        for (int kt = 0; kt < ktiles; ++kt) {
            // ---- A fragment (16x32 f16), direct from global ----
            // half j<8:  k = kt*32 + hi*8 + j
            // half j>=8: k = kt*32 + 16 + hi*8 + (j-8)
            v16h a;
            if (kt < kfull) {
                const float* p = arow + kt * 32 + hi * 8;
                const float4 f0 = *(const float4*)(p);
                const float4 f1 = *(const float4*)(p + 4);
                const float4 f2 = *(const float4*)(p + 16);
                const float4 f3 = *(const float4*)(p + 20);
                a[0]  = (_Float16)f0.x; a[1]  = (_Float16)f0.y;
                a[2]  = (_Float16)f0.z; a[3]  = (_Float16)f0.w;
                a[4]  = (_Float16)f1.x; a[5]  = (_Float16)f1.y;
                a[6]  = (_Float16)f1.z; a[7]  = (_Float16)f1.w;
                a[8]  = (_Float16)f2.x; a[9]  = (_Float16)f2.y;
                a[10] = (_Float16)f2.z; a[11] = (_Float16)f2.w;
                a[12] = (_Float16)f3.x; a[13] = (_Float16)f3.y;
                a[14] = (_Float16)f3.z; a[15] = (_Float16)f3.w;
            } else {  // tail: clamped load + select, no divergence
#pragma unroll
                for (int j = 0; j < 16; ++j) {
                    const int k = kt * 32 + hi * 8 + j + ((j >= 8) ? 8 : 0);
                    const int kc = (k < K) ? k : (K - 1);
                    const float v = arow[kc];
                    a[j] = (k < K) ? (_Float16)v : (_Float16)0.0f;
                }
            }

            // ---- B fragments, double-buffered from prepped weights ----
            const _Float16* wb = Wp + ((size_t)kt * 4096) + lane * 16;
            if (kt + 1 < ktiles)
                __builtin_prefetch(wb + 4096, 0, 0);  // global_prefetch_b8
            v16h bf[2];
            load_bfrag(bf[0], wb);
#pragma unroll
            for (int c = 0; c < 8; ++c) {
                if (c < 7)
                    load_bfrag(bf[(c + 1) & 1], wb + (c + 1) * 512);
                acc[c] = __builtin_amdgcn_wmma_f32_16x16x32_f16(
                    false, a, false, bf[c & 1], (short)0, acc[c],
                    false, false);
            }
        }
    }

    // ---- epilogue: lane half hi, reg r -> M = hi*8 + r, N = m ----
#pragma unroll
    for (int c = 0; c < 8; ++c) {
        const int col = c * 16 + m;
        const float bv = bias[col];
#pragma unroll
        for (int r = 0; r < 8; ++r) {
            const long row = rowBase + wave * 16 + hi * 8 + r;
            float v = acc[c][r] + bv;
            if (MODE >= 1) v = fmaxf(v, 0.0f);
            if (MODE == 2) v = fmaxf(v + addb[row * 128 + col], 0.0f);
            out[row * 128 + col] = v;
        }
    }
}

// degree counts: degInDst[dst]++ , degOutSrc[src]++
__global__ __launch_bounds__(256) void deg_kernel(
    const int* __restrict__ src, const int* __restrict__ dst,
    float* __restrict__ degOutSrc, float* __restrict__ degInDst)
{
    const long e = (long)blockIdx.x * 256 + threadIdx.x;
    if (e >= N_EDGES) return;
    atomicAdd(&degInDst[dst[e]], 1.0f);
    atomicAdd(&degOutSrc[src[e]], 1.0f);
}

// agg[sidx[e]] += h[gidx[e]]  (4 cols per thread, float4 gather)
__global__ __launch_bounds__(256) void scatter_kernel(
    const float* __restrict__ h, const int* __restrict__ gidx,
    const int* __restrict__ sidx, float* __restrict__ agg)
{
    const long t = (long)blockIdx.x * 256 + threadIdx.x;
    const long e = t >> 5;
    const int cq = (int)(t & 31) * 4;
    if (e >= N_EDGES) return;
    const long gs = gidx[e], sd = sidx[e];
    const float4 v = *(const float4*)(h + gs * 128 + cq);
    float* o = agg + sd * 128 + cq;
    atomicAdd(o + 0, v.x); atomicAdd(o + 1, v.y);
    atomicAdd(o + 2, v.z); atomicAdd(o + 3, v.w);
}

// mean normalization for both aggregation buffers
__global__ __launch_bounds__(256) void norm2_kernel(
    float* __restrict__ aggU, const float* __restrict__ degU,
    float* __restrict__ aggD, const float* __restrict__ degD)
{
    const long t = (long)blockIdx.x * 256 + threadIdx.x;
    const long n = t >> 5;
    const int cq = (int)(t & 31) * 4;
    if (n >= N_NODES) return;
    const float iu = 1.0f / fmaxf(degU[n], 1.0f);
    const float id = 1.0f / fmaxf(degD[n], 1.0f);
    float4* pu = (float4*)(aggU + n * 128 + cq);
    float4* pd = (float4*)(aggD + n * 128 + cq);
    float4 u = *pu, d = *pd;
    u.x *= iu; u.y *= iu; u.z *= iu; u.w *= iu;
    d.x *= id; d.y *= id; d.z *= id; d.w *= id;
    *pu = u; *pd = d;
}

// per-column sum / sumsq -> stats[0..127]=sum, stats[128..255]=sumsq
__global__ __launch_bounds__(256) void bn_stats_kernel(
    const float* __restrict__ x, float* __restrict__ stats)
{
    __shared__ float sh[512];
    const int tid = threadIdx.x;
    const int col = tid & 127, half = tid >> 7;
    const long rowBase = (long)blockIdx.x * 128;
    float s = 0.0f, q = 0.0f;
    for (int r = half; r < 128; r += 2) {
        const float v = x[(rowBase + r) * 128 + col];
        s += v; q += v * v;
    }
    sh[tid] = s; sh[256 + tid] = q;
    __syncthreads();
    if (tid < 128) {
        atomicAdd(&stats[col],       sh[tid] + sh[tid + 128]);
        atomicAdd(&stats[128 + col], sh[256 + tid] + sh[256 + tid + 128]);
    }
}

// h = relu( (x-mu)*rsqrt(var+eps)*gamma + beta ) + h   (residual in place)
__global__ __launch_bounds__(256) void bn_apply_kernel(
    const float* __restrict__ xm, float* __restrict__ h,
    const float* __restrict__ stats,
    const float* __restrict__ gamma, const float* __restrict__ beta)
{
    const long t = (long)blockIdx.x * 256 + threadIdx.x;
    const long n = t >> 5;
    const int cq = (int)(t & 31) * 4;
    if (n >= N_NODES) return;
    const float invN = 1.0f / (float)N_NODES;
    const float* xp = xm + n * 128 + cq;
    float* hp = h + n * 128 + cq;
#pragma unroll
    for (int j = 0; j < 4; ++j) {
        const int col = cq + j;
        const float mu  = stats[col] * invN;
        const float var = stats[128 + col] * invN - mu * mu;
        const float sc  = gamma[col] * __frsqrt_rn(var + BN_EPS);
        const float sf  = beta[col] - mu * sc;
        const float v = fmaxf(xp[j] * sc + sf, 0.0f);
        hp[j] = v + hp[j];
    }
}

// out[n, 0..1] = h[n,:] @ headW + headb   (C=2)
__global__ __launch_bounds__(256) void head_kernel(
    const float* __restrict__ h, const float* __restrict__ W,
    const float* __restrict__ b, float* __restrict__ out)
{
    const long n = (long)blockIdx.x * 256 + threadIdx.x;
    if (n >= N_NODES) return;
    float a0 = b[0], a1 = b[1];
    const float* hp = h + n * 128;
#pragma unroll 4
    for (int k = 0; k < 128; ++k) {
        const float v = hp[k];
        a0 = fmaf(v, W[k * 2 + 0], a0);
        a1 = fmaf(v, W[k * 2 + 1], a1);
    }
    out[n * 2 + 0] = a0;
    out[n * 2 + 1] = a1;
}

extern "C" void kernel_launch(void* const* d_in, const int* in_sizes, int n_in,
                              void* d_out, int out_size, void* d_ws, size_t ws_size,
                              hipStream_t stream)
{
    const int*   x_tag     = (const int*)  d_in[0];
    const float* x_text    = (const float*)d_in[1];
    const float* x_class   = (const float*)d_in[2];
    const float* x_num     = (const float*)d_in[3];
    const int*   edges     = (const int*)  d_in[4];
    const float* tag_embed = (const float*)d_in[5];
    const float* Wc   = (const float*)d_in[6];
    const float* bc   = (const float*)d_in[7];
    const float* Wtc  = (const float*)d_in[8];
    const float* btc  = (const float*)d_in[9];
    const float* Wt   = (const float*)d_in[10];
    const float* bt   = (const float*)d_in[11];
    const float* Wm   = (const float*)d_in[12];
    const float* bm   = (const float*)d_in[13];
    const float* Wn   = (const float*)d_in[14];
    const float* bnum = (const float*)d_in[15];
    const float* upWl = (const float*)d_in[16];
    const float* upbl = (const float*)d_in[17];
    const float* upWr = (const float*)d_in[18];
    const float* dnWl = (const float*)d_in[19];
    const float* dnbl = (const float*)d_in[20];
    const float* dnWr = (const float*)d_in[21];
    const float* pW   = (const float*)d_in[22];
    const float* pb   = (const float*)d_in[23];
    const float* gamma = (const float*)d_in[24];
    const float* beta  = (const float*)d_in[25];
    const float* headW = (const float*)d_in[26];
    const float* headb = (const float*)d_in[27];

    const int* src = edges;              // edge_index[0]
    const int* dst = edges + N_EDGES;    // edge_index[1]

    const size_t NB = (size_t)N_NODES * 128;
    float* B0 = (float*)d_ws;      // h (residual / layer state)
    float* B1 = B0 + NB;           // e_class  / aggU
    float* B2 = B1 + NB;           // h_tc     / aggD
    float* B3 = B2 + NB;           // h_text   / h_up
    float* B4 = B3 + NB;           // h_num    / h_down
    float* B5 = B4 + NB;           // h_merged
    float* degU  = B5 + NB;        // in-degree (dst)
    float* degD  = degU + N_NODES; // out-degree (src)
    float* stats = degD + N_NODES; // 256 floats: col sum / sumsq

    const dim3 blk(256);
    const dim3 gGemm(N_NODES / 128);
    const dim3 gEdge((N_EDGES + 255) / 256);
    const dim3 gEdge32((size_t)N_EDGES * 32 / 256);
    const dim3 gNode32((size_t)N_NODES * 32 / 256);
    const dim3 gNode((N_NODES + 255) / 256);

    // ---- one-time (per launch) weight prep into f16 fragment layout ----
    _Float16* wcur = (_Float16*)(stats + 256);  // 16B-aligned
    auto prep = [&](const float* W, int K) -> _Float16* {
        const int total = ((K + 31) >> 5) * 4096;
        prep_weight_kernel<<<dim3((total + 255) / 256), blk, 0, stream>>>(
            W, K, wcur, total);
        _Float16* ret = wcur;
        wcur += total;
        return ret;
    };
    _Float16* WnP   = prep(Wn, NUMF);
    _Float16* WcP   = prep(Wc, FT);
    _Float16* WtP   = prep(Wt, FT);
    _Float16* WtcTP = prep(Wtc, HD);
    _Float16* WtcBP = prep(Wtc + 128 * 128, HD);
    _Float16* WmTP  = prep(Wm, HD);
    _Float16* WmBP  = prep(Wm + 128 * 128, HD);
    _Float16 *upWlP[NLAYERS], *upWrP[NLAYERS], *dnWlP[NLAYERS],
             *dnWrP[NLAYERS], *pWTP[NLAYERS], *pWBP[NLAYERS];
    for (int i = 0; i < NLAYERS; ++i) {
        upWlP[i] = prep(upWl + (size_t)i * 128 * 128, HD);
        upWrP[i] = prep(upWr + (size_t)i * 128 * 128, HD);
        dnWlP[i] = prep(dnWl + (size_t)i * 128 * 128, HD);
        dnWrP[i] = prep(dnWr + (size_t)i * 128 * 128, HD);
        pWTP[i]  = prep(pW + (size_t)i * 256 * 128, HD);
        pWBP[i]  = prep(pW + (size_t)i * 256 * 128 + 128 * 128, HD);
    }

    // ---- Node encoder ----
    // h_num = relu(x_num @ Wn + bnum)            (K=16, tail-only tile)
    gemm128_wmma<1, false, false><<<gGemm, blk, 0, stream>>>(
        x_num, nullptr, WnP, NUMF, nullptr, nullptr, 0, bnum, nullptr, B4);
    // e_class = x_class @ Wc + bc                (K=300)
    gemm128_wmma<0, false, false><<<gGemm, blk, 0, stream>>>(
        x_class, nullptr, WcP, FT, nullptr, nullptr, 0, bc, nullptr, B1);
    // h_text = relu(x_text @ Wt + bt)            (K=300)
    gemm128_wmma<1, false, false><<<gGemm, blk, 0, stream>>>(
        x_text, nullptr, WtP, FT, nullptr, nullptr, 0, bt, nullptr, B3);
    // h_tc = relu(tag_embed[x_tag] @ WtcTop + e_class @ WtcBot + btc)
    gemm128_wmma<1, true, true><<<gGemm, blk, 0, stream>>>(
        tag_embed, x_tag, WtcTP, HD, B1, WtcBP, HD, btc, nullptr, B2);
    // h = relu( relu(h_tc @ WmTop + h_text @ WmBot + bm) + h_num )
    gemm128_wmma<2, true, false><<<gGemm, blk, 0, stream>>>(
        B2, nullptr, WmTP, HD, B3, WmBP, HD, bm, B4, B0);

    // ---- degrees (shared across layers) ----
    hipMemsetAsync(degU, 0, N_NODES * sizeof(float), stream);
    hipMemsetAsync(degD, 0, N_NODES * sizeof(float), stream);
    deg_kernel<<<gEdge, blk, 0, stream>>>(src, dst, degD, degU);

    // ---- SAGE layers ----
    for (int i = 0; i < NLAYERS; ++i) {
        hipMemsetAsync(B1, 0, NB * sizeof(float), stream);
        hipMemsetAsync(B2, 0, NB * sizeof(float), stream);
        scatter_kernel<<<gEdge32, blk, 0, stream>>>(B0, src, dst, B1); // up
        scatter_kernel<<<gEdge32, blk, 0, stream>>>(B0, dst, src, B2); // down
        norm2_kernel<<<gNode32, blk, 0, stream>>>(B1, degU, B2, degD);

        // h_up = aggU @ upWl + h @ upWr + upbl
        gemm128_wmma<0, true, false><<<gGemm, blk, 0, stream>>>(
            B1, nullptr, upWlP[i], HD, B0, upWrP[i], HD,
            upbl + i * 128, nullptr, B3);
        // h_down = aggD @ dnWl + h @ dnWr + dnbl
        gemm128_wmma<0, true, false><<<gGemm, blk, 0, stream>>>(
            B2, nullptr, dnWlP[i], HD, B0, dnWrP[i], HD,
            dnbl + i * 128, nullptr, B4);
        // h_merged = h_up @ pWTop + h_down @ pWBot + pb
        gemm128_wmma<0, true, false><<<gGemm, blk, 0, stream>>>(
            B3, nullptr, pWTP[i], HD, B4, pWBP[i], HD,
            pb + i * 128, nullptr, B5);

        hipMemsetAsync(stats, 0, 256 * sizeof(float), stream);
        bn_stats_kernel<<<dim3(N_NODES / 128), blk, 0, stream>>>(B5, stats);
        bn_apply_kernel<<<gNode32, blk, 0, stream>>>(B5, B0, stats,
                                                     gamma + i * 128,
                                                     beta + i * 128);
    }

    // ---- head ----
    head_kernel<<<gNode, blk, 0, stream>>>(B0, headW, headb, (float*)d_out);
}
